// INSA_Transformer_8581344657867
// MI455X (gfx1250) — matrix operand order repaired and checked by reading, not compiled, not run on hardware
//
#include <hip/hip_runtime.h>
#include <hip/hip_bf16.h>
#include <math.h>

// ---------------------------------------------------------------------------
// INSA transformer block pair for MI455X (gfx1250), bf16 WMMA everywhere.
// B=8, H=W=128, NW=8, D=256  ->  M = 131072 tokens, 512 windows of N=256.
// GEMM operands staged through LDS as bf16; pure-copy stagings use the CDNA5
// async global->LDS path (ASYNCcnt) when the toolchain exposes it.
// ---------------------------------------------------------------------------

typedef __attribute__((ext_vector_type(16))) __bf16 v16bf;
typedef __attribute__((ext_vector_type(8)))  float  v8f;
typedef int v2i __attribute__((ext_vector_type(2)));

union BF16x16 { v16bf v; unsigned short u[16]; };
union F32x8   { v8f   v; float          f[8];  };

__device__ __forceinline__ unsigned short f2bf(float f) {
  unsigned u = __float_as_uint(f);
  u += 0x7fffu + ((u >> 16) & 1u);          // round-to-nearest-even
  return (unsigned short)(u >> 16);
}
__device__ __forceinline__ float bf2f(unsigned short h) {
  return __uint_as_float(((unsigned)h) << 16);
}
__device__ __forceinline__ unsigned pk2(float lo, float hi) {
  return (unsigned)f2bf(lo) | ((unsigned)f2bf(hi) << 16);
}

// ---- CDNA5 async global->LDS copy (tracked by ASYNCcnt) -------------------
#if defined(__AMDGCN__) && __has_builtin(__builtin_amdgcn_global_load_async_to_lds_b64)
#define HAVE_ASYNC_LDS 1
#else
#define HAVE_ASYNC_LDS 0
#endif

__device__ __forceinline__ void g2l_b64(void* lds, const void* g) {
#if HAVE_ASYNC_LDS
  typedef v2i __attribute__((address_space(1))) *gp_t;   // global v2i*
  typedef v2i __attribute__((address_space(3))) *lp_t;   // LDS v2i*
  __builtin_amdgcn_global_load_async_to_lds_b64(
      (gp_t)(unsigned long long)g,
      (lp_t)(unsigned)(unsigned long long)lds, 0, 0);
#else
  *(uint2*)lds = *(const uint2*)g;
#endif
}
__device__ __forceinline__ void g2l_wait() {
#if HAVE_ASYNC_LDS
#if __has_builtin(__builtin_amdgcn_s_wait_asynccnt)
  __builtin_amdgcn_s_wait_asynccnt(0);
#else
  asm volatile("s_wait_asynccnt 0" ::: "memory");
#endif
#endif
}

// roll(-8,-8) + window partition index map (self-inverse for the scatter)
__device__ __forceinline__ int win_row_map(int bimg, int wy, int wx, int n) {
  int iy = n >> 4, ix = n & 15;
  int y = (wy * 16 + iy + 8) & 127;
  int x = (wx * 16 + ix + 8) & 127;
  return bimg * 16384 + y * 128 + x;
}

// ---------------------------------------------------------------------------
// Weight pre-pass: f32 [K][N] row-major  ->  bf16 W^T [N][K] row-major.
// ---------------------------------------------------------------------------
__global__ __launch_bounds__(256)
void cvt_transpose(const float* __restrict__ in, unsigned short* __restrict__ out,
                   int K, int N)
{
  size_t idx = (size_t)blockIdx.x * 256 + threadIdx.x;
  int k = (int)(idx / N), n = (int)(idx % N);
  out[(size_t)n * K + k] = f2bf(in[idx]);
}

// ---------------------------------------------------------------------------
// Generic bf16-WMMA GEMM: C[M,N] = op_a(A[M,K]) @ W[K,N],  W given as
// bf16 W^T[N][K].  Tile 128(M) x 64(N) x 32(K); 8 waves in 4x2; each wave a
// 32x32 patch = 2x2 v_wmma_f32_16x16x32_bf16.  M%128==0, N%64==0, K%32==0.
// ---------------------------------------------------------------------------
template<int ABF16, int AGELU, int CBF16>
__global__ __launch_bounds__(256)
void gemm_wmma(const void* __restrict__ Ap, const unsigned short* __restrict__ Wt,
               void* __restrict__ Cp, int M, int K, int N)
{
  __shared__ unsigned short As[128][40];   // [m][k]
  __shared__ unsigned short Ws[64][40];    // [n][k]  (transposed tile)

  const int tid  = threadIdx.x;
  const int lane = tid & 31, wv = tid >> 5;
  const int lm   = lane & 15, lh = lane >> 4;
  const int wm   = wv >> 1,   wn = wv & 1;
  const int m0   = blockIdx.y * 128;
  const int n0   = blockIdx.x * 64;

  F32x8 acc[2][2];
#pragma unroll
  for (int i = 0; i < 2; i++)
#pragma unroll
    for (int j = 0; j < 2; j++)
#pragma unroll
      for (int r = 0; r < 8; r++) acc[i][j].f[r] = 0.0f;

  const float*          Af = (const float*)Ap;
  const unsigned short* Ab = (const unsigned short*)Ap;

  for (int k0 = 0; k0 < K; k0 += 32) {
    if (k0 + 32 < K) {                       // prefetch next K tile
      if (tid < 128)
        __builtin_prefetch(ABF16 ? (const void*)(Ab + (size_t)(m0 + tid) * K + k0 + 32)
                                 : (const void*)(Af + (size_t)(m0 + tid) * K + k0 + 32), 0, 0);
      else if (tid < 192)
        __builtin_prefetch(Wt + (size_t)(n0 + tid - 128) * K + k0 + 32, 0, 0);
    }
    // stage W^T tile 64x32 first: pure bf16 copy -> async DMA to LDS
#pragma unroll
    for (int j = 0; j < 2; j++) {
      int i4 = tid + 256 * j;
      int r = i4 >> 3, c4 = (i4 & 7) * 4;
      g2l_b64(&Ws[r][c4], &Wt[(size_t)(n0 + r) * K + k0 + c4]);
    }
    // stage A tile 128x32
#pragma unroll
    for (int j = 0; j < 4; j++) {
      int i4 = tid + 256 * j;
      int r = i4 >> 3, c4 = (i4 & 7) * 4;
      if (ABF16 && !AGELU) {                 // pure copy -> async DMA
        g2l_b64(&As[r][c4], &Ab[(size_t)(m0 + r) * K + k0 + c4]);
      } else {
        uint2 p;
        if (ABF16) {
          p = *(const uint2*)&Ab[(size_t)(m0 + r) * K + k0 + c4];
          float x0 = bf2f((unsigned short)(p.x & 0xffff));
          float x1 = bf2f((unsigned short)(p.x >> 16));
          float x2 = bf2f((unsigned short)(p.y & 0xffff));
          float x3 = bf2f((unsigned short)(p.y >> 16));
          x0 = 0.5f * x0 * (1.0f + erff(x0 * 0.70710678f));
          x1 = 0.5f * x1 * (1.0f + erff(x1 * 0.70710678f));
          x2 = 0.5f * x2 * (1.0f + erff(x2 * 0.70710678f));
          x3 = 0.5f * x3 * (1.0f + erff(x3 * 0.70710678f));
          p.x = pk2(x0, x1); p.y = pk2(x2, x3);
        } else {
          float4 x = *(const float4*)&Af[(size_t)(m0 + r) * K + k0 + c4];
          if (AGELU) {
            x.x = 0.5f * x.x * (1.0f + erff(x.x * 0.70710678f));
            x.y = 0.5f * x.y * (1.0f + erff(x.y * 0.70710678f));
            x.z = 0.5f * x.z * (1.0f + erff(x.z * 0.70710678f));
            x.w = 0.5f * x.w * (1.0f + erff(x.w * 0.70710678f));
          }
          p.x = pk2(x.x, x.y); p.y = pk2(x.z, x.w);
        }
        *(uint2*)&As[r][c4] = p;
      }
    }
    g2l_wait();
    __syncthreads();

    BF16x16 a[2], b[2];
#pragma unroll
    for (int mi = 0; mi < 2; mi++) {
      int row = wm * 32 + mi * 16 + lm;
#pragma unroll
      for (int e = 0; e < 16; e++) {              // A-frag: contiguous k runs
        int k = lh * 16 + e;
        a[mi].u[e] = As[row][k];
      }
    }
#pragma unroll
    for (int ni = 0; ni < 2; ni++) {
      int col = wn * 32 + ni * 16 + lm;
#pragma unroll
      for (int e = 0; e < 16; e++)                // B-frag: contiguous k runs
        b[ni].u[e] = Ws[col][lh * 16 + e];
    }
#pragma unroll
    for (int mi = 0; mi < 2; mi++)
#pragma unroll
      for (int ni = 0; ni < 2; ni++)
        acc[mi][ni].v = __builtin_amdgcn_wmma_f32_16x16x32_bf16(
            false, a[mi].v, false, b[ni].v, (short)0, acc[mi][ni].v, false, false);
    __syncthreads();
  }

  float*          Cf = (float*)Cp;
  unsigned short* Cb = (unsigned short*)Cp;
#pragma unroll
  for (int mi = 0; mi < 2; mi++)
#pragma unroll
    for (int ni = 0; ni < 2; ni++) {
      int col = n0 + wn * 32 + ni * 16 + lm;
#pragma unroll
      for (int r = 0; r < 8; r++) {               // C/D: M = r + 8*lh
        int row = m0 + wm * 32 + mi * 16 + lh * 8 + r;
        if (CBF16) Cb[(size_t)row * N + col] = f2bf(acc[mi][ni].f[r]);
        else       Cf[(size_t)row * N + col] = acc[mi][ni].f[r];
      }
    }
}

// ---------------------------------------------------------------------------
// Windowed attention scores + softmax: one block = (window, 32-row chunk).
// S[32,256] = gather(Q) @ gather(K)^T / 16 + mask[widx]; softmax rows; P bf16.
// ---------------------------------------------------------------------------
__global__ __launch_bounds__(256)
void attn_scores(const float* __restrict__ Q, const float* __restrict__ Kx,
                 const float* __restrict__ mask, unsigned short* __restrict__ P)
{
  __shared__ unsigned short Qs[32][40];    // [q-row][c]
  __shared__ unsigned short Ks[256][40];   // [k-row][c]  (== B^T: contiguous)
  __shared__ float red[32][8];
  __shared__ float rowfin[32];

  const int tid  = threadIdx.x;
  const int lane = tid & 31, wv = tid >> 5;
  const int lm   = lane & 15, lh = lane >> 4;
  const int window = blockIdx.x >> 3;
  const int r0     = (blockIdx.x & 7) * 32;
  const int bimg = window >> 6, widx = window & 63;
  const int wy = widx >> 3, wx = widx & 7;

  F32x8 acc[2][2];
#pragma unroll
  for (int i = 0; i < 2; i++)
#pragma unroll
    for (int j = 0; j < 2; j++)
#pragma unroll
      for (int r = 0; r < 8; r++) acc[i][j].f[r] = 0.0f;

  for (int c0 = 0; c0 < 256; c0 += 32) {
    {                                             // Q chunk 32x32: 1 float4/thread
      int r = tid >> 3, c4 = (tid & 7) * 4;
      int g = win_row_map(bimg, wy, wx, r0 + r);
      float4 x = *(const float4*)&Q[(size_t)g * 256 + c0 + c4];
      uint2 p; p.x = pk2(x.x, x.y); p.y = pk2(x.z, x.w);
      *(uint2*)&Qs[r][c4] = p;
    }
#pragma unroll
    for (int j = 0; j < 8; j++) {                 // K tile 256x32: 8 float4/thread
      int i4 = tid + 256 * j;
      int r = i4 >> 3, c4 = (i4 & 7) * 4;
      int g = win_row_map(bimg, wy, wx, r);
      float4 x = *(const float4*)&Kx[(size_t)g * 256 + c0 + c4];
      uint2 p; p.x = pk2(x.x, x.y); p.y = pk2(x.z, x.w);
      *(uint2*)&Ks[r][c4] = p;
    }
    __syncthreads();

    BF16x16 a[2], b[2];
#pragma unroll
    for (int mi = 0; mi < 2; mi++) {
      int row = mi * 16 + lm;
#pragma unroll
      for (int e = 0; e < 16; e++) {
        int k = lh * 16 + e;
        a[mi].u[e] = Qs[row][k];
      }
    }
#pragma unroll
    for (int ni = 0; ni < 2; ni++) {
      int col = wv * 32 + ni * 16 + lm;           // S column = K row
#pragma unroll
      for (int e = 0; e < 16; e++)
        b[ni].u[e] = Ks[col][lh * 16 + e];
    }
#pragma unroll
    for (int mi = 0; mi < 2; mi++)
#pragma unroll
      for (int ni = 0; ni < 2; ni++)
        acc[mi][ni].v = __builtin_amdgcn_wmma_f32_16x16x32_bf16(
            false, a[mi].v, false, b[ni].v, (short)0, acc[mi][ni].v, false, false);
    __syncthreads();
  }

  // scale (1/sqrt(256)) + additive window mask
#pragma unroll
  for (int mi = 0; mi < 2; mi++)
#pragma unroll
    for (int ni = 0; ni < 2; ni++) {
      int col = wv * 32 + ni * 16 + lm;
#pragma unroll
      for (int r = 0; r < 8; r++) {
        int row = 16 * mi + 8 * lh + r;
        acc[mi][ni].f[r] = acc[mi][ni].f[r] * 0.0625f +
            mask[(size_t)widx * 65536 + (size_t)(r0 + row) * 256 + col];
      }
    }

  // row max: shfl-xor within 16-lane halves, then across waves via LDS
#pragma unroll
  for (int mi = 0; mi < 2; mi++)
#pragma unroll
    for (int r = 0; r < 8; r++) {
      float p = fmaxf(acc[mi][0].f[r], acc[mi][1].f[r]);
#pragma unroll
      for (int off = 1; off < 16; off <<= 1) p = fmaxf(p, __shfl_xor(p, off, 32));
      if (lm == 0) red[16 * mi + 8 * lh + r][wv] = p;
    }
  __syncthreads();
  if (tid < 32) {
    float m = red[tid][0];
#pragma unroll
    for (int w = 1; w < 8; w++) m = fmaxf(m, red[tid][w]);
    rowfin[tid] = m;
  }
  __syncthreads();

  // exp + row sum
#pragma unroll
  for (int mi = 0; mi < 2; mi++)
#pragma unroll
    for (int r = 0; r < 8; r++) {
      float m  = rowfin[16 * mi + 8 * lh + r];
      float e0 = __expf(acc[mi][0].f[r] - m); acc[mi][0].f[r] = e0;
      float e1 = __expf(acc[mi][1].f[r] - m); acc[mi][1].f[r] = e1;
      float p  = e0 + e1;
#pragma unroll
      for (int off = 1; off < 16; off <<= 1) p += __shfl_xor(p, off, 32);
      if (lm == 0) red[16 * mi + 8 * lh + r][wv] = p;
    }
  __syncthreads();
  if (tid < 32) {
    float s = 0.0f;
#pragma unroll
    for (int w = 0; w < 8; w++) s += red[tid][w];
    rowfin[tid] = s;
  }
  __syncthreads();

  // normalize, store P as bf16 (window-contiguous layout)
#pragma unroll
  for (int mi = 0; mi < 2; mi++)
#pragma unroll
    for (int ni = 0; ni < 2; ni++) {
      int col = wv * 32 + ni * 16 + lm;
#pragma unroll
      for (int r = 0; r < 8; r++) {
        int row = 16 * mi + 8 * lh + r;
        float inv = 1.0f / rowfin[row];
        P[(size_t)window * 65536 + (size_t)(r0 + row) * 256 + col] =
            f2bf(acc[mi][ni].f[r] * inv);
      }
    }
}

// ---------------------------------------------------------------------------
// out = P @ gather(V), scattered back through the inverse roll/merge map.
// P tile: async DMA copy; V staged transposed ([c][m]) for b128 B-frag reads.
// ---------------------------------------------------------------------------
__global__ __launch_bounds__(256)
void attn_pv(const unsigned short* __restrict__ P, const float* __restrict__ V,
             float* __restrict__ Out)
{
  __shared__ unsigned short Ps[32][40];     // [n][m-k]
  __shared__ unsigned short Vst[256][40];   // [c][m-k]  (transposed tile)

  const int tid  = threadIdx.x;
  const int lane = tid & 31, wv = tid >> 5;
  const int lm   = lane & 15, lh = lane >> 4;
  const int window = blockIdx.x >> 3;
  const int r0     = (blockIdx.x & 7) * 32;
  const int bimg = window >> 6, widx = window & 63;
  const int wy = widx >> 3, wx = widx & 7;

  F32x8 acc[2][2];
#pragma unroll
  for (int i = 0; i < 2; i++)
#pragma unroll
    for (int j = 0; j < 2; j++)
#pragma unroll
      for (int r = 0; r < 8; r++) acc[i][j].f[r] = 0.0f;

  for (int k0 = 0; k0 < 256; k0 += 32) {
    {                                             // P chunk 32x32 (bf16): async DMA
      int r = tid >> 3, c4 = (tid & 7) * 4;
      g2l_b64(&Ps[r][c4],
              &P[(size_t)window * 65536 + (size_t)(r0 + r) * 256 + k0 + c4]);
    }
#pragma unroll
    for (int j = 0; j < 8; j++) {                 // V tile 32x256, transpose-on-store
      int i4 = tid + 256 * j;
      int m = i4 >> 6, c4 = (i4 & 63) * 4;
      int g = win_row_map(bimg, wy, wx, k0 + m);
      float4 x = *(const float4*)&V[(size_t)g * 256 + c4];
      Vst[c4 + 0][m] = f2bf(x.x);
      Vst[c4 + 1][m] = f2bf(x.y);
      Vst[c4 + 2][m] = f2bf(x.z);
      Vst[c4 + 3][m] = f2bf(x.w);
    }
    g2l_wait();
    __syncthreads();

    BF16x16 a[2], b[2];
#pragma unroll
    for (int mi = 0; mi < 2; mi++) {
      int row = mi * 16 + lm;
#pragma unroll
      for (int e = 0; e < 16; e++) {
        int k = lh * 16 + e;
        a[mi].u[e] = Ps[row][k];
      }
    }
#pragma unroll
    for (int ni = 0; ni < 2; ni++) {
      int col = wv * 32 + ni * 16 + lm;           // output channel
#pragma unroll
      for (int e = 0; e < 16; e++)
        b[ni].u[e] = Vst[col][lh * 16 + e];
    }
#pragma unroll
    for (int mi = 0; mi < 2; mi++)
#pragma unroll
      for (int ni = 0; ni < 2; ni++)
        acc[mi][ni].v = __builtin_amdgcn_wmma_f32_16x16x32_bf16(
            false, a[mi].v, false, b[ni].v, (short)0, acc[mi][ni].v, false, false);
    __syncthreads();
  }

#pragma unroll
  for (int mi = 0; mi < 2; mi++)
#pragma unroll
    for (int ni = 0; ni < 2; ni++) {
      int col = wv * 32 + ni * 16 + lm;
#pragma unroll
      for (int r = 0; r < 8; r++) {
        int row = 16 * mi + 8 * lh + r;
        int g = win_row_map(bimg, wy, wx, r0 + row);   // scatter = same map
        Out[(size_t)g * 256 + col] = acc[mi][ni].f[r];
      }
    }
}

// ---------------------------------------------------------------------------
// LayerNorm over last dim (256) with optional residual add. 1 wave per row.
// ---------------------------------------------------------------------------
__global__ __launch_bounds__(256)
void ln_res(const float* __restrict__ X, const float* __restrict__ gma,
            const float* __restrict__ bta, const float* __restrict__ resid,
            float* __restrict__ Out)
{
  const int lane = threadIdx.x & 31, wv = threadIdx.x >> 5;
  const size_t row = (size_t)blockIdx.x * 8 + wv;
  const int c0 = lane * 8;

  float4 xa = *(const float4*)&X[row * 256 + c0];
  float4 xb = *(const float4*)&X[row * 256 + c0 + 4];

  float s = xa.x + xa.y + xa.z + xa.w + xb.x + xb.y + xb.z + xb.w;
#pragma unroll
  for (int off = 1; off < 32; off <<= 1) s += __shfl_xor(s, off, 32);
  float mean = s * (1.0f / 256.0f);

  float v = 0.0f;
  {
    float d;
    d = xa.x - mean; v += d * d;  d = xa.y - mean; v += d * d;
    d = xa.z - mean; v += d * d;  d = xa.w - mean; v += d * d;
    d = xb.x - mean; v += d * d;  d = xb.y - mean; v += d * d;
    d = xb.z - mean; v += d * d;  d = xb.w - mean; v += d * d;
  }
#pragma unroll
  for (int off = 1; off < 32; off <<= 1) v += __shfl_xor(v, off, 32);
  float rstd = rsqrtf(v * (1.0f / 256.0f) + 1e-5f);

  float4 ga = *(const float4*)&gma[c0], gb = *(const float4*)&gma[c0 + 4];
  float4 ba = *(const float4*)&bta[c0], bb = *(const float4*)&bta[c0 + 4];
  float4 oa, ob;
  oa.x = (xa.x - mean) * rstd * ga.x + ba.x;
  oa.y = (xa.y - mean) * rstd * ga.y + ba.y;
  oa.z = (xa.z - mean) * rstd * ga.z + ba.z;
  oa.w = (xa.w - mean) * rstd * ga.w + ba.w;
  ob.x = (xb.x - mean) * rstd * gb.x + bb.x;
  ob.y = (xb.y - mean) * rstd * gb.y + bb.y;
  ob.z = (xb.z - mean) * rstd * gb.z + bb.z;
  ob.w = (xb.w - mean) * rstd * gb.w + bb.w;
  if (resid) {
    float4 ra = *(const float4*)&resid[row * 256 + c0];
    float4 rb = *(const float4*)&resid[row * 256 + c0 + 4];
    oa.x += ra.x; oa.y += ra.y; oa.z += ra.z; oa.w += ra.w;
    ob.x += rb.x; ob.y += rb.y; ob.z += rb.z; ob.w += rb.w;
  }
  *(float4*)&Out[row * 256 + c0]     = oa;
  *(float4*)&Out[row * 256 + c0 + 4] = ob;
}

// Hcat[r, 0:256] = bf16(f0[r]);  Hcat[r, 256:512] = bf16(a_ln[r])
__global__ __launch_bounds__(256)
void concat_bf16(const float* __restrict__ A, const float* __restrict__ B,
                 unsigned short* __restrict__ Out)
{
  size_t i4 = (size_t)blockIdx.x * 256 + threadIdx.x;
  size_t row = i4 >> 7;
  int    c4  = (int)(i4 & 127) * 4;
  float4 x = (c4 < 256) ? *(const float4*)&A[row * 256 + c4]
                        : *(const float4*)&B[row * 256 + (c4 - 256)];
  uint2 p; p.x = pk2(x.x, x.y); p.y = pk2(x.z, x.w);
  *(uint2*)&Out[i4 * 4] = p;
}

// ---------------------------------------------------------------------------
extern "C" void kernel_launch(void* const* d_in, const int* in_sizes, int n_in,
                              void* d_out, int out_size, void* d_ws, size_t ws_size,
                              hipStream_t stream) {
  (void)in_sizes; (void)n_in; (void)out_size; (void)ws_size;

  const float* feat0 = (const float*)d_in[0];
  const float* feat1 = (const float*)d_in[1];
  const float* mask  = (const float*)d_in[2];
  const float* wq1   = (const float*)d_in[6];
  const float* wk1   = (const float*)d_in[7];
  const float* wv1   = (const float*)d_in[8];
  const float* wfc1  = (const float*)d_in[9];
  const float* g1    = (const float*)d_in[10];
  const float* b1    = (const float*)d_in[11];
  const float* wq2   = (const float*)d_in[12];
  const float* wk2   = (const float*)d_in[13];
  const float* wv2   = (const float*)d_in[14];
  const float* wfc2  = (const float*)d_in[15];
  const float* g2    = (const float*)d_in[16];
  const float* b2    = (const float*)d_in[17];
  const float* w1    = (const float*)d_in[18];
  const float* w2    = (const float*)d_in[19];
  const float* fg    = (const float*)d_in[20];
  const float* fb    = (const float*)d_in[21];

  const int M = 131072;                       // B * H * W
  char* ws = (char*)d_ws;
  float*          Xq   = (float*)(ws);
  float*          Xk   = (float*)(ws + 134217728ull);
  float*          Xv   = (float*)(ws + 268435456ull);
  float*          Xa   = (float*)(ws + 402653184ull);
  float*          Xf0  = (float*)(ws + 536870912ull);
  unsigned short* Pws  = (unsigned short*)(ws + 671088640ull);  //  67 MB bf16
  unsigned short* Hb   = (unsigned short*)(ws + 738197504ull);  // 536 MB bf16
  unsigned short* Hcat = (unsigned short*)Xq;                   // reuse Xq region
  unsigned short* Wtb  = (unsigned short*)(ws + 1275068416ull); //   4 MB bf16 W^T

  unsigned short* wb[8];
  const float* wsrc[8] = { wq1, wk1, wv1, wfc1, wq2, wk2, wv2, wfc2 };
  for (int i = 0; i < 8; i++) wb[i] = Wtb + (size_t)i * 65536;
  unsigned short* w1b = Wtb + 524288;            // 512*2048
  unsigned short* w2b = Wtb + 524288 + 1048576;  // 2048*256

  dim3 blk(256);
  dim3 gD(256 / 64, M / 128);     // D x D GEMMs
  dim3 gW1(2048 / 64, M / 128);   // FFN up
  dim3 gW2(256 / 64, M / 128);    // FFN down

  // ---- weight pre-pass: f32 -> bf16 transposed -------------------------
  for (int i = 0; i < 8; i++)
    cvt_transpose<<<65536 / 256, blk, 0, stream>>>(wsrc[i], wb[i], 256, 256);
  cvt_transpose<<<1048576 / 256, blk, 0, stream>>>(w1, w1b, 512, 2048);
  cvt_transpose<<<524288 / 256, blk, 0, stream>>>(w2, w2b, 2048, 256);

  // ---- block 1 (intra): q,k,v from feat0 -------------------------------
  gemm_wmma<0,0,0><<<gD, blk, 0, stream>>>(feat0, wb[0], Xq, M, 256, 256);
  gemm_wmma<0,0,0><<<gD, blk, 0, stream>>>(feat0, wb[1], Xk, M, 256, 256);
  gemm_wmma<0,0,0><<<gD, blk, 0, stream>>>(feat0, wb[2], Xv, M, 256, 256);
  attn_scores<<<4096, blk, 0, stream>>>(Xq, Xk, mask, Pws);
  attn_pv    <<<4096, blk, 0, stream>>>(Pws, Xv, Xa);
  gemm_wmma<0,0,0><<<gD, blk, 0, stream>>>(Xa, wb[3], Xq, M, 256, 256);
  ln_res<<<M / 8, blk, 0, stream>>>(Xq, g1, b1, feat0, Xf0);   // f0 = feat0 + LN(fc)

  // ---- block 2 (inter): q from f0, k/v from feat1 ----------------------
  gemm_wmma<0,0,0><<<gD, blk, 0, stream>>>(Xf0,   wb[4], Xq, M, 256, 256);
  gemm_wmma<0,0,0><<<gD, blk, 0, stream>>>(feat1, wb[5], Xk, M, 256, 256);
  gemm_wmma<0,0,0><<<gD, blk, 0, stream>>>(feat1, wb[6], Xv, M, 256, 256);
  attn_scores<<<4096, blk, 0, stream>>>(Xq, Xk, mask, Pws);
  attn_pv    <<<4096, blk, 0, stream>>>(Pws, Xv, Xa);
  gemm_wmma<0,0,0><<<gD, blk, 0, stream>>>(Xa, wb[7], Xq, M, 256, 256);
  ln_res<<<M / 8, blk, 0, stream>>>(Xq, g2, b2, nullptr, Xa);  // a = LN(fc2)

  // ---- FFN: h = gelu([f0, a] @ w1); out = f0 + LN(h @ w2) --------------
  concat_bf16<<<65536, blk, 0, stream>>>(Xf0, Xa, Hcat);
  gemm_wmma<1,0,1><<<gW1, blk, 0, stream>>>(Hcat, w1b, Hb, M, 512, 2048);
  gemm_wmma<1,1,0><<<gW2, blk, 0, stream>>>(Hb,   w2b, Xk, M, 2048, 256);  // gelu on A
  ln_res<<<M / 8, blk, 0, stream>>>(Xk, fg, fb, Xf0, (float*)d_out);
}
